// StabilizedGIMANSystem_10514079941119
// MI455X (gfx1250) — matrix operand order repaired and checked by reading, not compiled, not run on hardware
//
#include <hip/hip_runtime.h>
#include <hip/hip_bf16.h>

typedef __attribute__((ext_vector_type(16))) _Float16 v16h;
typedef __attribute__((ext_vector_type(8)))  _Float16 v8h;
typedef __attribute__((ext_vector_type(8)))  float    v8f;
typedef __attribute__((ext_vector_type(4)))  float    f32x4;

#define WMMA_F16(c, a, b) \
  (c) = __builtin_amdgcn_wmma_f32_16x16x32_f16(false, (a), false, (b), (short)0, (c), false, false)

#define WAVES    4
// ---- LDS layout: [x16 per wave: 4*6144][scratch pool: 4*30720] ----------
#define X16_SIZE 6144        // x [3][16][64] f16 per wave
#define SCR_BASE (WAVES * X16_SIZE)            // 24576
#define SCR_SIZE 30720
#define SCR_SB   0           // h f16 / qkv f16 / y f32   (18432)
#define SCR_O16  18432       // attn-out / LN-out f16     (6144)
#define SCR_CMB  24576       // combined [16][64] f32     (4096)
#define SCR_Z1   28672       // z1 [16][32] f16           (1024)
#define SCR_Z2   29696       // z2 [16][16] f32           (1024)
#define SMEM_TOTAL (SCR_BASE + WAVES * SCR_SIZE)   // 147456
// 32KB weight staging buffer aliases the scratch pool (scratch idle in phase 1)
#define WSTG_OFF SCR_BASE
#define WSTG_ELEMS (64 * 256)

// ---- f16 weight cache layout in d_ws (offsets in halves) ----
#define WO_WSP   0           // [64][256]
#define WO_WGE   16384
#define WO_WTM   32768
#define WO_GATW  49152       // [64][64]
#define WO_INW   53248       // [192][64]
#define WO_OUTW  65536       // [64][64]
#define WO_P1W   69632       // [32][64]
#define WO_P2W   71680       // [16][32]
#define WO_TOTAL 72192

// A fragment (16x32, f16) from a row-major f32 matrix (global or LDS).
__device__ __forceinline__ v16h ldA_f32(const float* X, int ldx, int k0) {
  const int lane = threadIdx.x & 31;
  const float* p = X + (size_t)(lane & 15) * ldx + k0 + ((lane >> 4) << 3);
  v16h r;
#pragma unroll
  for (int g = 0; g < 2; ++g) {
    const float* q = p + g * 16;
#pragma unroll
    for (int i = 0; i < 8; i += 4) {
      f32x4 t = *(const f32x4*)(q + i);
      r[g * 8 + i + 0] = (_Float16)t[0];
      r[g * 8 + i + 1] = (_Float16)t[1];
      r[g * 8 + i + 2] = (_Float16)t[2];
      r[g * 8 + i + 3] = (_Float16)t[3];
    }
  }
  return r;
}

// A fragment from a row-major f16 buffer (LDS).
__device__ __forceinline__ v16h ldA_s16(const _Float16* X, int ldx, int k0) {
  const int lane = threadIdx.x & 31;
  const _Float16* p = X + (lane & 15) * ldx + k0 + ((lane >> 4) << 3);
  v8h g0 = *(const v8h*)p;
  v8h g1 = *(const v8h*)(p + 16);
  return __builtin_shufflevector(g0, g1, 0,1,2,3,4,5,6,7,8,9,10,11,12,13,14,15);
}

// B fragment (32x16): B[k][n] = W[n][k], W row-major [N][K] f32 (fallback path).
__device__ __forceinline__ v16h ldB_f32(const float* W, int ldw, int n0, int k0) {
  const int lane = threadIdx.x & 31;
  const float* p = W + (size_t)(n0 + (lane & 15)) * ldw + k0 + ((lane >> 4) << 3);
  v16h r;
#pragma unroll
  for (int i = 0; i < 16; i += 4) {
    f32x4 t = *(const f32x4*)(p + (i < 8 ? i : i + 8));
    r[i + 0] = (_Float16)t[0];
    r[i + 1] = (_Float16)t[1];
    r[i + 2] = (_Float16)t[2];
    r[i + 3] = (_Float16)t[3];
  }
  return r;
}

// B fragment from f16 weights (global f16 cache or LDS stage): two b128 loads.
__device__ __forceinline__ v16h ldB_h16(const _Float16* W, int ldw, int n0, int k0) {
  const int lane = threadIdx.x & 31;
  const _Float16* p = W + (n0 + (lane & 15)) * ldw + k0 + ((lane >> 4) << 3);
  v8h g0 = *(const v8h*)p;
  v8h g1 = *(const v8h*)(p + 16);
  return __builtin_shufflevector(g0, g1, 0,1,2,3,4,5,6,7,8,9,10,11,12,13,14,15);
}

template <bool F16W>
__device__ __forceinline__ v16h ldB(const float* Wf, const _Float16* Wh,
                                    int ldw, int n0, int k0) {
  if constexpr (F16W) return ldB_h16(Wh, ldw, n0, k0);
  else                return ldB_f32(Wf, ldw, n0, k0);
}

// cooperative stage of one 64x256 weight matrix into LDS (f16)
template <bool F16W>
__device__ __forceinline__ void stage_w(_Float16* dst, const float* srcF,
                                        const _Float16* srcH) {
  const int t = threadIdx.x;  // 0..127
  if constexpr (F16W) {
    const v8h* s = (const v8h*)srcH;
    v8h* d = (v8h*)dst;
#pragma unroll
    for (int i = 0; i < WSTG_ELEMS / 8 / 128; ++i)  // 16 iters
      d[i * 128 + t] = s[i * 128 + t];
  } else {
#pragma unroll
    for (int i = 0; i < WSTG_ELEMS / 8 / 128; ++i) {
      int e = (i * 128 + t) * 8;
      f32x4 a = *(const f32x4*)(srcF + e);
      f32x4 b = *(const f32x4*)(srcF + e + 4);
      v8h o;
      o[0] = (_Float16)a[0]; o[1] = (_Float16)a[1];
      o[2] = (_Float16)a[2]; o[3] = (_Float16)a[3];
      o[4] = (_Float16)b[0]; o[5] = (_Float16)b[1];
      o[6] = (_Float16)b[2]; o[7] = (_Float16)b[3];
      *(v8h*)(dst + e) = o;
    }
  }
}

// one-shot f32 -> f16 weight conversion into d_ws
__global__ void cvt_weights(const float* __restrict__ s0, const float* __restrict__ s1,
                            const float* __restrict__ s2, const float* __restrict__ s3,
                            const float* __restrict__ s4, const float* __restrict__ s5,
                            const float* __restrict__ s6, const float* __restrict__ s7,
                            _Float16* __restrict__ dst) {
  int idx = blockIdx.x * 256 + threadIdx.x;
  if (idx >= WO_TOTAL) return;
  const int offs[9] = {WO_WSP, WO_WGE, WO_WTM, WO_GATW, WO_INW,
                       WO_OUTW, WO_P1W, WO_P2W, WO_TOTAL};
  const float* srcs[8] = {s0, s1, s2, s3, s4, s5, s6, s7};
  int seg = 0;
#pragma unroll
  for (int s = 0; s < 7; ++s)
    if (idx >= offs[s + 1]) seg = s + 1;
  dst[idx] = (_Float16)srcs[seg][idx - offs[seg]];
}

template <bool F16W>
__global__ __launch_bounds__(128, 1) void giman_fused(
    const float* __restrict__ spatial, const float* __restrict__ genomic,
    const float* __restrict__ temporal, const float* __restrict__ adj,
    const float* __restrict__ Wsp, const float* __restrict__ bsp,
    const float* __restrict__ Wge, const float* __restrict__ bge,
    const float* __restrict__ Wtm, const float* __restrict__ btm,
    const float* __restrict__ gatW, const float* __restrict__ gata,
    const float* __restrict__ inw, const float* __restrict__ inb,
    const float* __restrict__ outw, const float* __restrict__ outb,
    const float* __restrict__ lng, const float* __restrict__ lnb,
    const float* __restrict__ p1w, const float* __restrict__ p1b,
    const float* __restrict__ p2w, const float* __restrict__ p2b,
    const float* __restrict__ p3w, const float* __restrict__ p3b,
    const _Float16* __restrict__ wf16, float* __restrict__ out, int Btot) {
  extern __shared__ char smem[];
  const int lane  = threadIdx.x & 31;
  const int w     = threadIdx.x >> 5;
  const int nlo   = lane & 15;
  const int mbase = (lane >> 4) << 3;
  const int row0  = (blockIdx.x * WAVES + w) * 16;

  _Float16* x16 = (_Float16*)(smem + w * X16_SIZE);
  char* scr = smem + SCR_BASE + w * SCR_SIZE;
  _Float16* sb16 = (_Float16*)(scr + SCR_SB);
  float*    sbf  = (float*)(scr + SCR_SB);
  _Float16* o16  = (_Float16*)(scr + SCR_O16);
  float*    comb = (float*)(scr + SCR_CMB);
  _Float16* z1   = (_Float16*)(scr + SCR_Z1);
  float*    z2   = (float*)(scr + SCR_Z2);
  _Float16* wstg = (_Float16*)(smem + WSTG_OFF);  // aliases scratch (phase 1 only)

  const v8f vzero = {0.f, 0.f, 0.f, 0.f, 0.f, 0.f, 0.f, 0.f};

  // ---------------- Phase 1: modality projections (256->64) + ReLU --------
  {
    const float* srcs[3]  = {spatial, genomic, temporal};
    const float* Ws[3]    = {Wsp, Wge, Wtm};
    const int    Wo[3]    = {WO_WSP, WO_WGE, WO_WTM};
    const float* bs[3]    = {bsp, bge, btm};
#pragma unroll
    for (int m = 0; m < 3; ++m) {
      __syncthreads();  // scratch pool / previous stage no longer in use
      stage_w<F16W>(wstg, Ws[m], F16W ? (wf16 + Wo[m]) : (const _Float16*)nullptr);
      __syncthreads();
      const float* X = srcs[m] + (size_t)row0 * 256;
      v8f acc[4] = {vzero, vzero, vzero, vzero};
      for (int k0 = 0; k0 < 256; k0 += 32) {
        v16h a = ldA_f32(X, 256, k0);
#pragma unroll
        for (int nt = 0; nt < 4; ++nt) {
          v16h b = ldB_h16(wstg, 256, nt * 16, k0);   // from LDS stage
          WMMA_F16(acc[nt], a, b);
        }
      }
#pragma unroll
      for (int nt = 0; nt < 4; ++nt) {
        float bias = bs[m][nt * 16 + nlo];
#pragma unroll
        for (int j = 0; j < 8; ++j) {
          float v = acc[nt][j] + bias;
          v = v > 0.f ? v : 0.f;
          x16[(m * 16 + mbase + j) * 64 + nt * 16 + nlo] = (_Float16)v;
        }
      }
    }
  }
  __syncthreads();

  // ---------------- Phase 2: GAT: h = x @ gatW^T, attention, pooled -------
  {
    v16h b[4][2];  // gat_W fragments hoisted: shared by all 3 nodes
#pragma unroll
    for (int nt = 0; nt < 4; ++nt) {
      b[nt][0] = ldB<F16W>(gatW, wf16 + WO_GATW, 64, nt * 16, 0);
      b[nt][1] = ldB<F16W>(gatW, wf16 + WO_GATW, 64, nt * 16, 32);
    }
#pragma unroll
    for (int n = 0; n < 3; ++n) {
      v16h a0 = ldA_s16(x16 + n * 1024, 64, 0);
      v16h a1 = ldA_s16(x16 + n * 1024, 64, 32);
#pragma unroll
      for (int nt = 0; nt < 4; ++nt) {
        v8f c = vzero;
        WMMA_F16(c, a0, b[nt][0]);
        WMMA_F16(c, a1, b[nt][1]);
#pragma unroll
        for (int j = 0; j < 8; ++j)
          sb16[(n * 16 + mbase + j) * 64 + nt * 16 + nlo] = (_Float16)c[j];
      }
    }
  }
  __syncthreads();
  {
    const int head = lane >> 4, r = lane & 15;
    const _Float16* h0 = sb16 + (0 * 16 + r) * 64 + head * 32;
    const _Float16* h1 = sb16 + (1 * 16 + r) * 64 + head * 32;
    const _Float16* h2 = sb16 + (2 * 16 + r) * 64 + head * 32;
    const _Float16* hp[3] = {h0, h1, h2};
    float ei[3], ej[3];
#pragma unroll
    for (int n = 0; n < 3; ++n) {
      float si = 0.f, sj = 0.f;
      for (int d = 0; d < 32; ++d) {
        float hv = (float)hp[n][d];
        si = fmaf(hv, gata[d], si);
        sj = fmaf(hv, gata[32 + d], sj);
      }
      ei[n] = si; ej[n] = sj;
    }
    const float* ad = adj + (size_t)(row0 + r) * 9;
    float wj[3] = {0.f, 0.f, 0.f};
#pragma unroll
    for (int i = 0; i < 3; ++i) {
      float e[3], mx = -3.4e38f;
#pragma unroll
      for (int jx = 0; jx < 3; ++jx) {
        float v = ei[i] + ej[jx];
        v = v > 0.f ? v : 0.2f * v;           // LeakyReLU(0.2)
        if (ad[i * 3 + jx] == 0.f) v = -1e9f; // adjacency mask
        e[jx] = v;
        mx = fmaxf(mx, v);
      }
      float ex0 = __expf(e[0] - mx), ex1 = __expf(e[1] - mx), ex2 = __expf(e[2] - mx);
      float inv = 1.f / (ex0 + ex1 + ex2);
      wj[0] += ex0 * inv; wj[1] += ex1 * inv; wj[2] += ex2 * inv;
    }
    wj[0] *= (1.f / 3.f); wj[1] *= (1.f / 3.f); wj[2] *= (1.f / 3.f);
    float* cr = comb + r * 64 + head * 32;
    for (int d = 0; d < 32; ++d)
      cr[d] = wj[0] * (float)h0[d] + wj[1] * (float)h1[d] + wj[2] * (float)h2[d];
  }
  __syncthreads();

  // ---------------- Phase 3: MHA -----------------------------------------
  // qkv = x @ in_w^T + in_b -> sb16 as [3][16][192] f16 (A hoisted over nt)
  {
    v16h a[3][2];
#pragma unroll
    for (int n = 0; n < 3; ++n) {
      a[n][0] = ldA_s16(x16 + n * 1024, 64, 0);
      a[n][1] = ldA_s16(x16 + n * 1024, 64, 32);
    }
#pragma unroll
    for (int nt = 0; nt < 12; ++nt) {
      v16h b0 = ldB<F16W>(inw, wf16 + WO_INW, 64, nt * 16, 0);
      v16h b1 = ldB<F16W>(inw, wf16 + WO_INW, 64, nt * 16, 32);
      float bias = inb[nt * 16 + nlo];
#pragma unroll
      for (int n = 0; n < 3; ++n) {
        v8f c = vzero;
        WMMA_F16(c, a[n][0], b0);
        WMMA_F16(c, a[n][1], b1);
#pragma unroll
        for (int j = 0; j < 8; ++j)
          sb16[(n * 16 + mbase + j) * 192 + nt * 16 + nlo] = (_Float16)(c[j] + bias);
      }
    }
  }
  __syncthreads();
  {
    const int head = lane >> 4, r = lane & 15;
    float att[3][3];
#pragma unroll
    for (int i = 0; i < 3; ++i) {
      const _Float16* qi = sb16 + (i * 16 + r) * 192 + head * 32;
#pragma unroll
      for (int jx = 0; jx < 3; ++jx) {
        const _Float16* kj = sb16 + (jx * 16 + r) * 192 + 64 + head * 32;
        float s = 0.f;
        for (int d = 0; d < 32; ++d) s = fmaf((float)qi[d], (float)kj[d], s);
        att[i][jx] = s * 0.17677669529663687f; // 1/sqrt(32)
      }
    }
#pragma unroll
    for (int i = 0; i < 3; ++i) {
      float mx = fmaxf(att[i][0], fmaxf(att[i][1], att[i][2]));
      float e0 = __expf(att[i][0] - mx), e1 = __expf(att[i][1] - mx), e2 = __expf(att[i][2] - mx);
      float inv = 1.f / (e0 + e1 + e2);
      att[i][0] = e0 * inv; att[i][1] = e1 * inv; att[i][2] = e2 * inv;
    }
    const _Float16* v0 = sb16 + (0 * 16 + r) * 192 + 128 + head * 32;
    const _Float16* v1 = sb16 + (1 * 16 + r) * 192 + 128 + head * 32;
    const _Float16* v2 = sb16 + (2 * 16 + r) * 192 + 128 + head * 32;
#pragma unroll
    for (int i = 0; i < 3; ++i) {
      _Float16* op = o16 + (i * 16 + r) * 64 + head * 32;
      for (int d = 0; d < 32; ++d) {
        float vv = att[i][0] * (float)v0[d] + att[i][1] * (float)v1[d] + att[i][2] * (float)v2[d];
        op[d] = (_Float16)vv;
      }
    }
  }
  __syncthreads();
  // out-projection + residual: y = x + o @ out_w^T + out_b  -> sbf f32
  {
    v16h b[4][2];  // out_w fragments hoisted
#pragma unroll
    for (int nt = 0; nt < 4; ++nt) {
      b[nt][0] = ldB<F16W>(outw, wf16 + WO_OUTW, 64, nt * 16, 0);
      b[nt][1] = ldB<F16W>(outw, wf16 + WO_OUTW, 64, nt * 16, 32);
    }
#pragma unroll
    for (int n = 0; n < 3; ++n) {
      v16h a0 = ldA_s16(o16 + n * 1024, 64, 0);
      v16h a1 = ldA_s16(o16 + n * 1024, 64, 32);
#pragma unroll
      for (int nt = 0; nt < 4; ++nt) {
        v8f c = vzero;
        WMMA_F16(c, a0, b[nt][0]);
        WMMA_F16(c, a1, b[nt][1]);
        float bias = outb[nt * 16 + nlo];
#pragma unroll
        for (int j = 0; j < 8; ++j) {
          int idx = (n * 16 + mbase + j) * 64 + nt * 16 + nlo;
          sbf[idx] = (float)x16[idx] + c[j] + bias;
        }
      }
    }
  }
  __syncthreads();

  // ---------------- Phase 4: LayerNorm + cross-modal mean -----------------
  for (int it = 0; it < 2; ++it) {
    int p = it * 32 + lane;
    if (p < 48) {
      int n = p >> 4, r = p & 15;
      const float* yr = sbf + (n * 16 + r) * 64;
      float mu = 0.f;
      for (int d = 0; d < 64; ++d) mu += yr[d];
      mu *= (1.f / 64.f);
      float var = 0.f;
      for (int d = 0; d < 64; ++d) { float t = yr[d] - mu; var = fmaf(t, t, var); }
      var *= (1.f / 64.f);
      float rstd = rsqrtf(var + 1e-5f);
      _Float16* lo = o16 + (n * 16 + r) * 64;
      for (int d = 0; d < 64; ++d)
        lo[d] = (_Float16)(((yr[d] - mu) * rstd) * lng[d] + lnb[d]);
    }
  }
  __syncthreads();
  {
    const int r = lane & 15, hh = lane >> 4;
    for (int dd = 0; dd < 32; ++dd) {
      int d = hh * 32 + dd;
      float s = (float)o16[(0 * 16 + r) * 64 + d] + (float)o16[(1 * 16 + r) * 64 + d] +
                (float)o16[(2 * 16 + r) * 64 + d];
      comb[r * 64 + d] += s * (1.f / 3.f);
    }
  }
  __syncthreads();

  // ---------------- Phase 5: predictor 64->32->16->2 ----------------------
  {
    v16h a0 = ldA_f32(comb, 64, 0);   // comb lives in LDS; f32 loads + cvt
    v16h a1 = ldA_f32(comb, 64, 32);
#pragma unroll
    for (int nt = 0; nt < 2; ++nt) {
      v8f c = vzero;
      WMMA_F16(c, a0, ldB<F16W>(p1w, wf16 + WO_P1W, 64, nt * 16, 0));
      WMMA_F16(c, a1, ldB<F16W>(p1w, wf16 + WO_P1W, 64, nt * 16, 32));
      float bias = p1b[nt * 16 + nlo];
#pragma unroll
      for (int j = 0; j < 8; ++j) {
        float v = c[j] + bias;
        v = v > 0.f ? v : 0.f;
        z1[(mbase + j) * 32 + nt * 16 + nlo] = (_Float16)v;
      }
    }
  }
  __syncthreads();
  {
    v16h a = ldA_s16(z1, 32, 0);
    v8f c = vzero;
    WMMA_F16(c, a, ldB<F16W>(p2w, wf16 + WO_P2W, 32, 0, 0));
    float bias = p2b[nlo];
#pragma unroll
    for (int j = 0; j < 8; ++j) {
      float v = c[j] + bias;
      v = v > 0.f ? v : 0.f;
      z2[(mbase + j) * 16 + nlo] = v;
    }
  }
  __syncthreads();
  {
    const int r = lane & 15, cc = lane >> 4;  // 16 rows x 2 output cols
    float acc = p3b[cc];
    for (int k = 0; k < 16; ++k) acc = fmaf(z2[r * 16 + k], p3w[cc * 16 + k], acc);
    out[(size_t)cc * Btot + row0 + r] = acc;  // tuple order: col0 block, col1 block
  }
}

template <bool F16W>
static void launch_main(void* const* d_in, void* d_out, void* d_ws, int Btot,
                        hipStream_t stream) {
  const size_t shmem = SMEM_TOTAL;
  auto* kptr = giman_fused<F16W>;
  hipFuncSetAttribute(reinterpret_cast<const void*>(kptr),
                      hipFuncAttributeMaxDynamicSharedMemorySize, (int)shmem);
  giman_fused<F16W><<<dim3(Btot / (WAVES * 16)), dim3(128), shmem, stream>>>(
      (const float*)d_in[0],  (const float*)d_in[1],  (const float*)d_in[2],
      (const float*)d_in[3],  (const float*)d_in[4],  (const float*)d_in[5],
      (const float*)d_in[6],  (const float*)d_in[7],  (const float*)d_in[8],
      (const float*)d_in[9],  (const float*)d_in[10], (const float*)d_in[11],
      (const float*)d_in[12], (const float*)d_in[13], (const float*)d_in[14],
      (const float*)d_in[15], (const float*)d_in[16], (const float*)d_in[17],
      (const float*)d_in[18], (const float*)d_in[19], (const float*)d_in[20],
      (const float*)d_in[21], (const float*)d_in[22], (const float*)d_in[23],
      (const _Float16*)d_ws, (float*)d_out, Btot);
}

extern "C" void kernel_launch(void* const* d_in, const int* in_sizes, int n_in,
                              void* d_out, int out_size, void* d_ws, size_t ws_size,
                              hipStream_t stream) {
  (void)n_in; (void)out_size;
  const int Btot = in_sizes[0] / 256;  // 131072
  const bool use_f16w = (d_ws != nullptr) && (ws_size >= WO_TOTAL * sizeof(_Float16));

  if (use_f16w) {
    cvt_weights<<<dim3((WO_TOTAL + 255) / 256), dim3(256), 0, stream>>>(
        (const float*)d_in[4], (const float*)d_in[6], (const float*)d_in[8],
        (const float*)d_in[10], (const float*)d_in[12], (const float*)d_in[14],
        (const float*)d_in[18], (const float*)d_in[20], (_Float16*)d_ws);
    launch_main<true>(d_in, d_out, d_ws, Btot, stream);
  } else {
    launch_main<false>(d_in, d_out, nullptr, Btot, stream);
  }
}
